// DynamicHPF_257698038333
// MI455X (gfx1250) — matrix operand order repaired and checked by reading, not compiled, not run on hardware
//
#include <hip/hip_runtime.h>
#include <hip/hip_bf16.h>
#include <stdint.h>

typedef __attribute__((ext_vector_type(16))) __bf16 v16bf;
typedef __attribute__((ext_vector_type(8)))  float  v8f;

#define N_IMG 4
#define C_IN  128
#define H_IMG 192
#define W_IMG 192
#define HP    194   // padded
#define WP    194
#define C_OUT 18    // G*K*K
#define C_OUT_PAD 32
#define KTOT  1152  // 9 taps * 128 channels
#define EPS_BN 1e-5f

union Frag { v16bf v; uint4 u4[2]; };

__device__ __forceinline__ int refl(int i, int n) {
    // numpy/jax 'reflect' with pad=1: -1 -> 1, n -> n-2
    if (i < 0) i = -i;
    if (i >= n) i = 2 * n - 2 - i;
    return i;
}

__device__ __forceinline__ uint16_t f2bf(float f) {
    uint32_t u = __float_as_uint(f);
    uint32_t r = (u + 0x7FFFu + ((u >> 16) & 1u)) >> 16;  // RNE
    return (uint16_t)r;
}

// ---------------------------------------------------------------------------
// Kernel 1: NCHW f32 -> padded NHWC bf16 (reflect pad baked in).
// grid (4 wblocks, 194 rows, 4 images), block 256, LDS tile transpose.
// ---------------------------------------------------------------------------
__global__ void pack_x_kernel(const float* __restrict__ x, uint16_t* __restrict__ x_cl) {
    __shared__ float tile[32][65];
    const int n  = blockIdx.z;
    const int hp = blockIdx.y;
    const int w0 = blockIdx.x * 64;
    const int hs = refl(hp - 1, H_IMG);
    const int tx = threadIdx.x & 63;     // w within tile
    const int ty = threadIdx.x >> 6;     // 0..3

    for (int cb = 0; cb < 4; ++cb) {     // 4 blocks of 32 channels
        // coalesced read: fixed channel, consecutive w
        #pragma unroll
        for (int i = 0; i < 8; ++i) {
            const int cl = ty * 8 + i;
            const int c  = cb * 32 + cl;
            const int wp = w0 + tx;
            float val = 0.f;
            if (wp < WP) {
                const int ws = refl(wp - 1, W_IMG);
                val = x[(((size_t)n * C_IN + c) * H_IMG + hs) * W_IMG + ws];
            }
            tile[cl][tx] = val;
        }
        __syncthreads();
        // packed write: 16 bytes (8 bf16 channels) per thread, contiguous in c
        const int wl = threadIdx.x >> 2;     // 0..63
        const int cq = threadIdx.x & 3;      // 0..3
        const int wp = w0 + wl;
        if (wp < WP) {
            uint4 v;
            uint32_t pk[4];
            #pragma unroll
            for (int j = 0; j < 4; ++j) {
                const float f0 = tile[cq * 8 + 2 * j + 0][wl];
                const float f1 = tile[cq * 8 + 2 * j + 1][wl];
                pk[j] = (uint32_t)f2bf(f0) | ((uint32_t)f2bf(f1) << 16);
            }
            v.x = pk[0]; v.y = pk[1]; v.z = pk[2]; v.w = pk[3];
            *(uint4*)(x_cl + ((((size_t)n * HP + hp) * WP + wp) * C_IN + cb * 32 + cq * 8)) = v;
        }
        __syncthreads();
    }
}

// ---------------------------------------------------------------------------
// Kernel 2: conv_w (18,128,3,3) f32 -> w_pack (32,1152) bf16, k = tap*128 + c
// ---------------------------------------------------------------------------
__global__ void pack_w_kernel(const float* __restrict__ cw, uint16_t* __restrict__ w_pack) {
    for (int idx = threadIdx.x; idx < C_OUT_PAD * KTOT; idx += blockDim.x) {
        const int m = idx / KTOT;
        const int k = idx % KTOT;
        const int t = k >> 7;
        const int c = k & 127;
        float v = 0.f;
        if (m < C_OUT) {
            const int dy = t / 3, dx = t % 3;
            v = cw[(((size_t)m * C_IN + c) * 3 + dy) * 3 + dx];
        }
        w_pack[idx] = f2bf(v);
    }
}

// ---------------------------------------------------------------------------
// Kernel 3: sigma = softmax(BN(conv)) - center, via bf16 WMMA implicit GEMM.
// One wave per 16-pixel tile; block = 4 waves (64 pixels of one row).
// grid (3, 192, 4), block 128. A = weights (2 M-tiles), B = patch, K=1152.
// ---------------------------------------------------------------------------
__global__ void sigma_kernel(const uint16_t* __restrict__ x_cl,
                             const uint16_t* __restrict__ w_pack,
                             const float* __restrict__ bn_gamma,
                             const float* __restrict__ bn_beta,
                             const float* __restrict__ bn_mean,
                             const float* __restrict__ bn_var,
                             float* __restrict__ sigma_out) {
    __shared__ float sig[4][16 * 32];

    const int n    = blockIdx.z;
    const int h    = blockIdx.y;
    const int wave = threadIdx.x >> 5;
    const int lane = threadIdx.x & 31;
    const int w0   = blockIdx.x * 64 + wave * 16;

    const int mrow = lane & 15;   // A row (out channel) / D column (pixel)
    const int half = lane >> 4;

    const uint16_t* arow0 = w_pack + (size_t)mrow * KTOT;
    const uint16_t* arow1 = w_pack + (size_t)(mrow + 16) * KTOT;

    v8f acc0 = {};
    v8f acc1 = {};

    #pragma unroll
    for (int t = 0; t < 9; ++t) {
        const int dy = t / 3, dx = t % 3;
        // B fragment row base for this lane's pixel (contiguous 128 channels)
        const uint16_t* brow =
            x_cl + (((size_t)n * HP + (h + dy)) * WP + (w0 + mrow + dx)) * C_IN;
        #pragma unroll
        for (int cb = 0; cb < 4; ++cb) {
            const int k0 = t * 128 + cb * 32;
            // B (32x16, 16-bit): lane holds k = e + 16*half, contiguous run of 16
            Frag b;
            b.u4[0] = *(const uint4*)(brow + cb * 32 + 16 * half + 0);
            b.u4[1] = *(const uint4*)(brow + cb * 32 + 16 * half + 8);
            // A (16x32, 16-bit): lane holds k = e + 16*half, contiguous run of 16
            Frag a0, a1;
            a0.u4[0] = *(const uint4*)(arow0 + k0 + 16 * half);
            a0.u4[1] = *(const uint4*)(arow0 + k0 + 16 * half + 8);
            a1.u4[0] = *(const uint4*)(arow1 + k0 + 16 * half);
            a1.u4[1] = *(const uint4*)(arow1 + k0 + 16 * half + 8);

            acc0 = __builtin_amdgcn_wmma_f32_16x16x32_bf16(
                false, a0.v, false, b.v, (short)0, acc0, false, false);
            acc1 = __builtin_amdgcn_wmma_f32_16x16x32_bf16(
                false, a1.v, false, b.v, (short)0, acc1, false, false);
        }
    }

    // D layout: VGPR r, pixel = lane%16, channel = r + 8*half (tile1: +16)
    float* s = sig[wave];
    #pragma unroll
    for (int r = 0; r < 8; ++r) {
        s[mrow * 32 + (r + 8 * half)]      = acc0[r];
        s[mrow * 32 + 16 + (r + 8 * half)] = acc1[r];
    }
    __syncthreads();

    // Epilogue: lane (pix, g) = (lane&15, lane>>4): BN fold + softmax(9) - center
    const int pix = lane & 15;
    const int g   = half;
    float v[9];
    float mx = -1e30f;
    #pragma unroll
    for (int j = 0; j < 9; ++j) {
        const int ch = g * 9 + j;
        const float inv = bn_gamma[ch] * rsqrtf(bn_var[ch] + EPS_BN);
        const float val = s[pix * 32 + ch] * inv + (bn_beta[ch] - bn_mean[ch] * inv);
        v[j] = val;
        mx = fmaxf(mx, val);
    }
    float sum = 0.f;
    #pragma unroll
    for (int j = 0; j < 9; ++j) { v[j] = __expf(v[j] - mx); sum += v[j]; }
    const float rs = 1.f / sum;
    float* op = sigma_out + (((size_t)n * H_IMG + h) * W_IMG + (w0 + pix)) * C_OUT + g * 9;
    #pragma unroll
    for (int j = 0; j < 9; ++j) {
        op[j] = v[j] * rs - (j == 4 ? 1.f : 0.f);
    }
}

// ---------------------------------------------------------------------------
// Kernel 4: out[n,c,h,w] = sum_t sigma[n,h,w,g*9+t] * x[n,c,refl(h-1+dy),refl(w-1+dx)]
// ---------------------------------------------------------------------------
__global__ void apply_kernel(const float* __restrict__ x,
                             const float* __restrict__ sigma,
                             float* __restrict__ out) {
    const size_t idx = (size_t)blockIdx.x * blockDim.x + threadIdx.x;
    const int w = (int)(idx % W_IMG);
    size_t r = idx / W_IMG;
    const int h = (int)(r % H_IMG); r /= H_IMG;
    const int c = (int)(r % C_IN);
    const int n = (int)(r / C_IN);
    const int g = c >> 6;

    const float* sp = sigma + (((size_t)n * H_IMG + h) * W_IMG + w) * C_OUT + g * 9;
    const float* xb = x + ((size_t)n * C_IN + c) * H_IMG * W_IMG;

    float acc = 0.f;
    #pragma unroll
    for (int t = 0; t < 9; ++t) {
        const int dy = t / 3, dx = t % 3;
        const int hh = refl(h - 1 + dy, H_IMG);
        const int ww = refl(w - 1 + dx, W_IMG);
        acc = fmaf(sp[t], xb[(size_t)hh * W_IMG + ww], acc);
    }
    out[idx] = acc;
}

// ---------------------------------------------------------------------------
extern "C" void kernel_launch(void* const* d_in, const int* in_sizes, int n_in,
                              void* d_out, int out_size, void* d_ws, size_t ws_size,
                              hipStream_t stream) {
    const float* x        = (const float*)d_in[0];
    const float* conv_w   = (const float*)d_in[1];
    const float* bn_gamma = (const float*)d_in[2];
    const float* bn_beta  = (const float*)d_in[3];
    const float* bn_mean  = (const float*)d_in[4];
    const float* bn_var   = (const float*)d_in[5];
    float* out = (float*)d_out;

    // workspace layout
    const size_t x_cl_elems = (size_t)N_IMG * HP * WP * C_IN;      // bf16
    uint16_t* x_cl   = (uint16_t*)d_ws;
    uint16_t* w_pack = x_cl + x_cl_elems;                          // 16B-aligned
    float*    sigma  = (float*)(w_pack + (size_t)C_OUT_PAD * KTOT);

    dim3 gpx(4, HP, N_IMG);
    pack_x_kernel<<<gpx, 256, 0, stream>>>(x, x_cl);

    pack_w_kernel<<<1, 256, 0, stream>>>(conv_w, w_pack);

    dim3 gs(W_IMG / 64, H_IMG, N_IMG);
    sigma_kernel<<<gs, 128, 0, stream>>>(x_cl, w_pack,
                                         bn_gamma, bn_beta, bn_mean, bn_var,
                                         sigma);

    const size_t total = (size_t)N_IMG * C_IN * H_IMG * W_IMG;
    apply_kernel<<<(unsigned)(total / 256), 256, 0, stream>>>(x, sigma, out);
}
